// TransformerLayerShard_31842887532575
// MI455X (gfx1250) — compile-verified, matrix operand order checked by reading, not run on hardware
//
#include <hip/hip_runtime.h>
#include <hip/hip_bf16.h>
#include <math.h>

typedef _Float16 half_t;
typedef _Float16 v8h  __attribute__((ext_vector_type(8)));
typedef _Float16 v16h __attribute__((ext_vector_type(16)));
typedef float    v8f  __attribute__((ext_vector_type(8)));

#define T_DIM 2048
#define D_DIM 2048
#define H_DIM 16
#define DH_DIM 128
#define FF_DIM 8192

#define BM 128
#define BN 64
#define BK 32

// ---------------------------------------------------------------------------
// Async Global->LDS 16-byte copy (CDNA5, tracked by ASYNCcnt).
// vdst = 32-bit LDS byte address, vaddr = 32-bit byte offset, saddr = base.
// ---------------------------------------------------------------------------
__device__ __forceinline__ void async_ld16(unsigned int lds_addr,
                                           unsigned int goff,
                                           unsigned long long base) {
  asm volatile("global_load_async_to_lds_b128 %0, %1, %2"
               :: "v"(lds_addr), "v"(goff), "s"(base)
               : "memory");
}

// ---------------------------------------------------------------------------
// Fragment loader for the CDNA5 16x16x32 f16 WMMA "A"/"B" VGPR layout,
// reading straight from global NT-layout (row-major [rows][ld], contraction
// contiguous). Used by the attention kernel.
// ---------------------------------------------------------------------------
__device__ __forceinline__ v16h load_frag_nt(const half_t* __restrict__ base,
                                             int row, int k0, int ld) {
  const int lane = threadIdx.x & 31;
  const int r    = lane & 15;
  const int hf   = lane >> 4;
  const half_t* p = base + (size_t)(row + r) * ld + k0 + 8 * hf;
  union { v16h v; v8h h[2]; } u;
  u.h[0] = *(const v8h*)(p);
  u.h[1] = *(const v8h*)(p + 16);
  return u.v;
}

// ---------------------------------------------------------------------------
// LayerNorm: one block per row, fp32 stats, f16 output (WMMA A operand).
// ---------------------------------------------------------------------------
__global__ void ln_kernel(const float* __restrict__ x,
                          const float* __restrict__ scale,
                          const float* __restrict__ offset,
                          half_t* __restrict__ xh) {
  __shared__ float s1[256], s2[256];
  const int t = blockIdx.x, tid = threadIdx.x;
  const float* xr = x + (size_t)t * D_DIM;
  float sum = 0.f, sq = 0.f;
  for (int d = tid; d < D_DIM; d += 256) { float v = xr[d]; sum += v; sq += v * v; }
  s1[tid] = sum; s2[tid] = sq;
  __syncthreads();
  for (int s = 128; s > 0; s >>= 1) {
    if (tid < s) { s1[tid] += s1[tid + s]; s2[tid] += s2[tid + s]; }
    __syncthreads();
  }
  const float mean = s1[0] * (1.0f / D_DIM);
  const float var  = s2[0] * (1.0f / D_DIM) - mean * mean;
  const float inv  = rsqrtf(var + 1e-5f);
  for (int d = tid; d < D_DIM; d += 256)
    xh[(size_t)t * D_DIM + d] = (half_t)(scale[d] * inv * (xr[d] - mean) + offset[d]);
}

// ---------------------------------------------------------------------------
// fp32 W[K][N]  ->  f16 W^T[N][K]
// ---------------------------------------------------------------------------
__global__ void cvt_t_kernel(const float* __restrict__ w, half_t* __restrict__ wt,
                             int K, int N) {
  size_t idx = (size_t)blockIdx.x * blockDim.x + threadIdx.x;
  if (idx >= (size_t)K * N) return;
  size_t k = idx / N, n = idx % N;
  wt[n * (size_t)K + k] = (half_t)w[idx];
}

// ---------------------------------------------------------------------------
// Rotary embedding on first 64 dims of every head, in-place on f16 q/k.
// ---------------------------------------------------------------------------
__global__ void rotary_kernel(half_t* __restrict__ q, half_t* __restrict__ k) {
  int idx = blockIdx.x * blockDim.x + threadIdx.x;
  if (idx >= T_DIM * H_DIM * 32) return;
  const int i = idx & 31;
  const int h = (idx >> 5) & (H_DIM - 1);
  const int t = idx >> 9;
  const float inv = __expf(-(2.0f * (float)i / 64.0f) * logf(10000.0f));
  const float ang = (float)t * inv;
  const float c = __cosf(ang), s = __sinf(ang);
  const size_t base = (size_t)t * D_DIM + h * DH_DIM + 2 * i;
  float q1 = (float)q[base], q2 = (float)q[base + 1];
  q[base]     = (half_t)(q1 * c - q2 * s);
  q[base + 1] = (half_t)(q2 * c + q1 * s);
  float k1 = (float)k[base], k2 = (float)k[base + 1];
  k[base]     = (half_t)(k1 * c - k2 * s);
  k[base + 1] = (half_t)(k2 * c + k1 * s);
}

// ---------------------------------------------------------------------------
// WMMA GEMM with async Global->LDS double buffering.
//   C[M][N](fp32 acc) = A[M][K](f16) x (Bt[N][K])^T(f16)
// Block = 256 thr / 8 waves; tile 128x64x32. Per k-step each thread issues
// 3 async b128 copies (A: 2, B: 1) into the next buffer, then waves compute
// 4 WMMAs each from ds_load_b128 fragments. B tile shared by all 8 waves.
// Epilogue: optional bias + tanh-GELU; fp32 / f16 / f16-transposed outputs.
// ---------------------------------------------------------------------------
__global__ void __launch_bounds__(256)
gemm_nt_async_kernel(const half_t* __restrict__ A,
                     const half_t* __restrict__ Bt,
                     const float* __restrict__ bias,
                     float* __restrict__ outF,
                     half_t* __restrict__ outH,
                     half_t* __restrict__ outHT,
                     int M, int N, int K, int do_gelu) {
  __shared__ half_t smA[2][BM * BK];   // row-major [row][k]
  __shared__ half_t smB[2][BN * BK];   // row-major [n][k]

  const int tid  = threadIdx.x;
  const int lane = tid & 31;
  const int wv   = tid >> 5;              // wave 0..7 -> rows wv*16..+15
  const int r = lane & 15, hf = lane >> 4;

  const int nb = blockIdx.x;              // N / 64
  const int mb = blockIdx.y;              // M / 128

  // --- async copy addressing (per thread) --------------------------------
  // A tile: 512 chunks of 16B; thread handles chunks tid and tid+256.
  // B tile: 256 chunks of 16B; thread handles chunk tid.
  const int ca0 = tid, ca1 = tid + 256;
  const unsigned int gA0 =
      (unsigned int)((((long long)mb * BM + (ca0 >> 2)) * K + (ca0 & 3) * 8) * 2);
  const unsigned int gA1 =
      (unsigned int)((((long long)mb * BM + (ca1 >> 2)) * K + (ca1 & 3) * 8) * 2);
  const unsigned int gB =
      (unsigned int)((((long long)nb * BN + (tid >> 2)) * K + (tid & 3) * 8) * 2);

  unsigned int lA[2], lB[2];
  lA[0] = (unsigned int)(uintptr_t)(&smA[0][0]) + (unsigned int)ca0 * 16u;
  lA[1] = (unsigned int)(uintptr_t)(&smA[1][0]) + (unsigned int)ca0 * 16u;
  // second A chunk sits 256*16 bytes further in the same buffer
  lB[0] = (unsigned int)(uintptr_t)(&smB[0][0]) + (unsigned int)tid * 16u;
  lB[1] = (unsigned int)(uintptr_t)(&smB[1][0]) + (unsigned int)tid * 16u;

  const unsigned long long baseA = (unsigned long long)(uintptr_t)A;
  const unsigned long long baseB = (unsigned long long)(uintptr_t)Bt;

  auto issue = [&](int buf, int kb) {
    const unsigned int ko = (unsigned int)kb * 2u;
    async_ld16(lA[buf],             gA0 + ko, baseA);
    async_ld16(lA[buf] + 256u*16u,  gA1 + ko, baseA);
    async_ld16(lB[buf],             gB  + ko, baseB);
  };

  v8f acc[4] = {};
  issue(0, 0);
  int buf = 0;
  for (int kb = 0; kb < K; kb += BK) {
    const bool more = (kb + BK) < K;
    if (more) {
      issue(buf ^ 1, kb + BK);
      asm volatile("s_wait_asynccnt 0x3" ::: "memory");  // this buffer done
    } else {
      asm volatile("s_wait_asynccnt 0x0" ::: "memory");
    }
    __syncthreads();   // all waves' async data visible

    union { v16h v; v8h h[2]; } ua;
    const int arow = (wv * 16 + r) * BK;
    ua.h[0] = *(const v8h*)&smA[buf][arow + 8 * hf];
    ua.h[1] = *(const v8h*)&smA[buf][arow + 16 + 8 * hf];
#pragma unroll
    for (int j = 0; j < 4; ++j) {
      union { v16h v; v8h h[2]; } ub;
      const int brow = (j * 16 + r) * BK;
      ub.h[0] = *(const v8h*)&smB[buf][brow + 8 * hf];
      ub.h[1] = *(const v8h*)&smB[buf][brow + 16 + 8 * hf];
      acc[j] = __builtin_amdgcn_wmma_f32_16x16x32_f16(
          false, ua.v, false, ub.v, (short)0, acc[j], false, false);
    }
    __syncthreads();   // reads done before buffer is overwritten
    buf ^= 1;
  }

  // --- epilogue -----------------------------------------------------------
  const int m0 = mb * BM + wv * 16 + 8 * hf;  // lane holds rows m0..m0+7
#pragma unroll
  for (int j = 0; j < 4; ++j) {
    const int n = nb * BN + j * 16 + r;
    const float bv = bias ? bias[n] : 0.0f;
    float vals[8];
#pragma unroll
    for (int i = 0; i < 8; ++i) {
      float v = acc[j][i] + bv;
      if (do_gelu) {
        float u = 0.7978845608028654f * (v + 0.044715f * v * v * v);
        v = 0.5f * v * (1.0f + tanhf(u));
      }
      vals[i] = v;
    }
    if (outF) {
#pragma unroll
      for (int i = 0; i < 8; ++i) outF[(size_t)(m0 + i) * N + n] = vals[i];
    }
    if (outH) {
#pragma unroll
      for (int i = 0; i < 8; ++i) outH[(size_t)(m0 + i) * N + n] = (half_t)vals[i];
    }
    if (outHT) {  // transposed f16: rows contiguous -> single b128 store
      v8h hv;
#pragma unroll
      for (int i = 0; i < 8; ++i) hv[i] = (half_t)vals[i];
      *(v8h*)(outHT + (size_t)n * M + m0) = hv;
    }
  }
}

// ---------------------------------------------------------------------------
// Flash attention, one wave per (16-query tile, head).
// q,k: f16 [T][D]; vT: f16 [D][T] (pre-transposed); ctx: f16 [T][D].
// Online softmax in fp32; P re-layout C-format -> A-fragment via 1KB LDS.
// ---------------------------------------------------------------------------
__global__ void attn_kernel(const half_t* __restrict__ q,
                            const half_t* __restrict__ k,
                            const half_t* __restrict__ vT,
                            half_t* __restrict__ ctx,
                            const float* __restrict__ attn_bias) {
  __shared__ half_t pbuf[16 * 32];
  const int lane = threadIdx.x;
  const int r = lane & 15, hf = lane >> 4;
  const int qt = blockIdx.x, h = blockIdx.y;
  const float scale = 0.08838834764831845f;  // 1/sqrt(128)
  const float bias = attn_bias[0];

  v16h aq[4];
#pragma unroll
  for (int kc = 0; kc < 4; ++kc)
    aq[kc] = load_frag_nt(q, qt * 16, h * DH_DIM + kc * 32, D_DIM);

  float mi[8], li[8];
  v8f o[8] = {};
#pragma unroll
  for (int i = 0; i < 8; ++i) { mi[i] = -1e30f; li[i] = 0.0f; }

  const int nblk = ((qt * 16 + 15) >> 5) + 1;  // causal
  for (int jb = 0; jb < nblk; ++jb) {
    const int key0 = jb * 32;
    v8f s[2] = {};
#pragma unroll
    for (int sub = 0; sub < 2; ++sub)
#pragma unroll
      for (int kc = 0; kc < 4; ++kc) {
        v16h bk = load_frag_nt(k, key0 + sub * 16, h * DH_DIM + kc * 32, D_DIM);
        s[sub] = __builtin_amdgcn_wmma_f32_16x16x32_f16(
            false, aq[kc], false, bk, (short)0, s[sub], false, false);
      }

    float alpha[8];
#pragma unroll
    for (int i = 0; i < 8; ++i) {
      const int row = qt * 16 + i + 8 * hf;
      float v0 = s[0][i] * scale + bias;
      float v1 = s[1][i] * scale + bias;
      if (key0 + r > row)      v0 = -1e9f;
      if (key0 + 16 + r > row) v1 = -1e9f;
      float m = fmaxf(v0, v1);
#pragma unroll
      for (int off = 1; off < 16; off <<= 1)
        m = fmaxf(m, __shfl_xor(m, off, 32));
      const float mnew = fmaxf(mi[i], m);
      alpha[i] = __expf(mi[i] - mnew);
      const float p0 = __expf(v0 - mnew);
      const float p1 = __expf(v1 - mnew);
      s[0][i] = p0; s[1][i] = p1;
      float ls = p0 + p1;
#pragma unroll
      for (int off = 1; off < 16; off <<= 1)
        ls += __shfl_xor(ls, off, 32);
      li[i] = li[i] * alpha[i] + ls;
      mi[i] = mnew;
    }
#pragma unroll
    for (int t8 = 0; t8 < 8; ++t8)
#pragma unroll
      for (int i = 0; i < 8; ++i) o[t8][i] *= alpha[i];

    __syncthreads();
#pragma unroll
    for (int sub = 0; sub < 2; ++sub)
#pragma unroll
      for (int i = 0; i < 8; ++i)
        pbuf[(i + 8 * hf) * 32 + sub * 16 + r] = (half_t)s[sub][i];
    __syncthreads();

    union { v16h v; v8h hh[2]; } pa;
    pa.hh[0] = *(const v8h*)(&pbuf[r * 32 + 8 * hf]);
    pa.hh[1] = *(const v8h*)(&pbuf[r * 32 + 16 + 8 * hf]);

#pragma unroll
    for (int t8 = 0; t8 < 8; ++t8) {
      v16h bv = load_frag_nt(vT, h * DH_DIM + t8 * 16, key0, T_DIM);
      o[t8] = __builtin_amdgcn_wmma_f32_16x16x32_f16(
          false, pa.v, false, bv, (short)0, o[t8], false, false);
    }
  }

#pragma unroll
  for (int i = 0; i < 8; ++i) li[i] = 1.0f / li[i];
#pragma unroll
  for (int t8 = 0; t8 < 8; ++t8) {
    const int n = h * DH_DIM + t8 * 16 + r;
#pragma unroll
    for (int i = 0; i < 8; ++i)
      ctx[(size_t)(qt * 16 + i + 8 * hf) * D_DIM + n] = (half_t)(o[t8][i] * li[i]);
  }
}

// ---------------------------------------------------------------------------
__global__ void add3_kernel(const float* __restrict__ x,
                            const float* __restrict__ a,
                            const float* __restrict__ f,
                            float* __restrict__ out, int n) {
  int i = blockIdx.x * blockDim.x + threadIdx.x;
  if (i < n) out[i] = x[i] + a[i] + f[i];
}

// ---------------------------------------------------------------------------
extern "C" void kernel_launch(void* const* d_in, const int* in_sizes, int n_in,
                              void* d_out, int out_size, void* d_ws, size_t ws_size,
                              hipStream_t stream) {
  const float* x      = (const float*)d_in[0];
  const float* abias  = (const float*)d_in[1];
  const float* scale  = (const float*)d_in[2];
  const float* offset = (const float*)d_in[3];
  const float* wq = (const float*)d_in[4];
  const float* wk = (const float*)d_in[5];
  const float* wv = (const float*)d_in[6];
  const float* wo = (const float*)d_in[7];
  const float* w1 = (const float*)d_in[8];
  const float* b1 = (const float*)d_in[9];
  const float* w2 = (const float*)d_in[10];
  const float* b2 = (const float*)d_in[11];
  float* out = (float*)d_out;
  (void)in_sizes; (void)n_in; (void)out_size; (void)ws_size;

  char* p = (char*)d_ws;
  auto alloc = [&](size_t bytes) -> char* {
    char* r = p;
    p += (bytes + 255) & ~(size_t)255;
    return r;
  };
  const size_t TD = (size_t)T_DIM * D_DIM;
  const size_t DD = (size_t)D_DIM * D_DIM;
  const size_t DF = (size_t)D_DIM * FF_DIM;

  half_t* xh   = (half_t*)alloc(TD * 2);
  half_t* wqT  = (half_t*)alloc(DD * 2);
  half_t* wkT  = (half_t*)alloc(DD * 2);
  half_t* wvT  = (half_t*)alloc(DD * 2);
  half_t* woT  = (half_t*)alloc(DD * 2);
  half_t* w1T  = (half_t*)alloc(DF * 2);
  half_t* w2T  = (half_t*)alloc(DF * 2);
  half_t* qb   = (half_t*)alloc(TD * 2);
  half_t* kb   = (half_t*)alloc(TD * 2);
  half_t* vTb  = (half_t*)alloc(TD * 2);
  half_t* ctx  = (half_t*)alloc(TD * 2);
  half_t* h1b  = (half_t*)alloc((size_t)T_DIM * FF_DIM * 2);
  float*  attnO = (float*)alloc(TD * 4);
  float*  ffO   = (float*)alloc(TD * 4);

  // 1) LayerNorm -> f16
  ln_kernel<<<T_DIM, 256, 0, stream>>>(x, scale, offset, xh);

  // 2) Weight cast + transpose (f16, NT layout; all fit in 192MB L2)
  const int gDD = (int)((DD + 255) / 256);
  const int gDF = (int)((DF + 255) / 256);
  cvt_t_kernel<<<gDD, 256, 0, stream>>>(wq, wqT, D_DIM, D_DIM);
  cvt_t_kernel<<<gDD, 256, 0, stream>>>(wk, wkT, D_DIM, D_DIM);
  cvt_t_kernel<<<gDD, 256, 0, stream>>>(wv, wvT, D_DIM, D_DIM);
  cvt_t_kernel<<<gDD, 256, 0, stream>>>(wo, woT, D_DIM, D_DIM);
  cvt_t_kernel<<<gDF, 256, 0, stream>>>(w1, w1T, D_DIM, FF_DIM);
  cvt_t_kernel<<<gDF, 256, 0, stream>>>(w2, w2T, FF_DIM, D_DIM);

  // 3) QKV projections (V written pre-transposed for attention B-fragments)
  const dim3 gP(D_DIM / BN, T_DIM / BM);       // 32 x 16
  gemm_nt_async_kernel<<<gP, 256, 0, stream>>>(xh, wqT, nullptr, nullptr, qb, nullptr,
                                               T_DIM, D_DIM, D_DIM, 0);
  gemm_nt_async_kernel<<<gP, 256, 0, stream>>>(xh, wkT, nullptr, nullptr, kb, nullptr,
                                               T_DIM, D_DIM, D_DIM, 0);
  gemm_nt_async_kernel<<<gP, 256, 0, stream>>>(xh, wvT, nullptr, nullptr, nullptr, vTb,
                                               T_DIM, D_DIM, D_DIM, 0);

  // 4) Rotary on q,k
  rotary_kernel<<<(T_DIM * H_DIM * 32) / 256, 256, 0, stream>>>(qb, kb);

  // 5) Causal flash attention
  attn_kernel<<<dim3(T_DIM / 16, H_DIM), 32, 0, stream>>>(qb, kb, vTb, ctx, abias);

  // 6) Output projection
  gemm_nt_async_kernel<<<gP, 256, 0, stream>>>(ctx, woT, nullptr, attnO, nullptr, nullptr,
                                               T_DIM, D_DIM, D_DIM, 0);

  // 7) FFN
  const dim3 gF1(FF_DIM / BN, T_DIM / BM);     // 128 x 16
  gemm_nt_async_kernel<<<gF1, 256, 0, stream>>>(xh, w1T, b1, nullptr, h1b, nullptr,
                                                T_DIM, FF_DIM, D_DIM, 1);
  gemm_nt_async_kernel<<<gP, 256, 0, stream>>>(h1b, w2T, b2, ffO, nullptr, nullptr,
                                               T_DIM, D_DIM, FF_DIM, 0);

  // 8) Residual sum
  add3_kernel<<<(int)((TD + 255) / 256), 256, 0, stream>>>(x, attnO, ffO, out, (int)TD);
}
